// Impaint_15083925144092
// MI455X (gfx1250) — matrix operand order repaired and checked by reading, not compiled, not run on hardware
//
#include <hip/hip_runtime.h>
#include <type_traits>

typedef __attribute__((ext_vector_type(16))) _Float16 v16h;
typedef __attribute__((ext_vector_type(4)))  _Float16 v4h;
typedef __attribute__((ext_vector_type(8)))  float    v8f;
typedef __attribute__((ext_vector_type(4)))  float    v4f;

#define BATCH 8

__device__ __forceinline__ int lower_bound_i(const int* __restrict__ a, int n, int key) {
    int lo = 0, hi = n;
    while (lo < hi) { int mid = (lo + hi) >> 1; if (a[mid] < key) lo = mid + 1; else hi = mid; }
    return lo;
}

// y[b,v,f] = alpha * sum_{e: rows[e]==v} vals[e] * x[b, cols[e], f]  - (sub ? sub[b,v,f] : 0)
// One wave per node v; lanes cover the flattened B*(F/VW) vector-output space.
// VW=4 (float4 / global_load_b128) when F%4==0, else scalar.
template<int F>
__global__ void spmm_kernel(const int* __restrict__ rows, const int* __restrict__ cols,
                            const float* __restrict__ vals, int E, int V,
                            const float* __restrict__ x, const float* __restrict__ sub,
                            float alpha, float* __restrict__ y) {
    constexpr int VW   = (F % 4 == 0) ? 4 : 1;
    constexpr int FV   = F / VW;
    constexpr int TOTV = BATCH * FV;
    constexpr int PER  = (TOTV + 31) / 32;
    using vec_t = std::conditional_t<VW == 4, v4f, float>;

    const int wave = (blockIdx.x * blockDim.x + threadIdx.x) >> 5;
    const int lane = threadIdx.x & 31;
    if (wave >= V) return;
    const int v = wave;
    const int start = lower_bound_i(rows, E, v);
    const int end   = lower_bound_i(rows, E, v + 1);

    vec_t acc[PER] = {};
    for (int e = start; e < end; ++e) {
        const float w = vals[e];
        const int   c = cols[e];
        const vec_t* __restrict__ xv = (const vec_t*)x;
#pragma unroll
        for (int i = 0; i < PER; ++i) {
            const int idx = lane + 32 * i;
            if (idx < TOTV) {
                const int b = idx / FV, f4 = idx % FV;
                acc[i] += w * xv[((size_t)b * V + c) * FV + f4];
            }
        }
    }
    vec_t* __restrict__ yv = (vec_t*)y;
    const vec_t* __restrict__ sv = (const vec_t*)sub;
#pragma unroll
    for (int i = 0; i < PER; ++i) {
        const int idx = lane + 32 * i;
        if (idx < TOTV) {
            const int b = idx / FV, f4 = idx % FV;
            const size_t o = ((size_t)b * V + v) * FV + f4;
            vec_t r = alpha * acc[i];
            if (sub) r -= sv[o];
            yv[o] = r;
        }
    }
}

// Tall-skinny GEMM with ReLU: out[r, 0:FOUT] = relu([T0|T1|T2][r, :] @ Wh + bias)
// A = 16-row node tile (f16 via LDS), B = stacked/padded f16 weights, f32 accumulate via WMMA.
template<int FIN, int FOUT, int KSTEPS, int NTILES>
__global__ __launch_bounds__(128, 1)
void cheb_wmma_kernel(const float* __restrict__ T0, const float* __restrict__ T1,
                      const float* __restrict__ T2, const _Float16* __restrict__ Wh,
                      const float* __restrict__ bias, float* __restrict__ out,
                      int nrows) {
    constexpr int KPAD = KSTEPS * 32;
    constexpr int NWAVES = 4;
    constexpr int PADK = KPAD - 3 * FIN;              // zero-padded K tail
    static_assert(FIN % 8 == 0 && PADK >= 0, "staging assumptions");
    __shared__ _Float16 sW[KPAD * FOUT];
    __shared__ _Float16 sA[NWAVES][16 * KPAD];

    const int tid  = threadIdx.x;
    const int wid  = tid >> 5;
    const int lane = tid & 31;
    const int hi   = lane >> 4;   // 0 for lanes 0-15, 1 for lanes 16-31
    const int lrow = lane & 15;

    for (int i = tid; i < KPAD * FOUT / 4; i += (int)blockDim.x)
        ((v4h*)sW)[i] = ((const v4h*)Wh)[i];
    __syncthreads();

    // Build weight-stationary B fragments once.
    // B fragment, contiguous k order used here (same as A): element i, lane l -> K = i + 16*(l>=16), N = l&15.
    v16h Bf[KSTEPS * NTILES];
#pragma unroll
    for (int ks = 0; ks < KSTEPS; ++ks) {
#pragma unroll
        for (int nt = 0; nt < NTILES; ++nt) {
            v16h bfrag;
#pragma unroll
            for (int i = 0; i < 16; ++i) {
                const int k = ks * 32 + hi * 16 + i;
                const int n = nt * 16 + lrow;
                bfrag[i] = sW[k * FOUT + n];
            }
            Bf[ks * NTILES + nt] = bfrag;
        }
    }

    // Per-lane bias (column = nt*16 + lrow), loaded once.
    float bv[NTILES];
#pragma unroll
    for (int nt = 0; nt < NTILES; ++nt) bv[nt] = bias[nt * 16 + lrow];

    _Float16* sAw = &sA[wid][0];
    // Zero-fill the constant pad region of the A strip once (never rewritten).
    if constexpr (PADK > 0) {
#pragma unroll
        for (int it = 0; it < 16 * PADK / 4 / 32; ++it) {
            const int id  = lane + 32 * it;
            const int row = id / (PADK / 4);
            const int c4  = (id % (PADK / 4)) * 4;
            v4h z = {};
            *(v4h*)(&sAw[row * KPAD + 3 * FIN + c4]) = z;
        }
    }

    const float* Ts[3] = {T0, T1, T2};
    const int ntiles_rows = (nrows + 15) >> 4;
    for (int tile = blockIdx.x * NWAVES + wid; tile < ntiles_rows; tile += gridDim.x * NWAVES) {
        const bool full = (tile * 16 + 16) <= nrows;

        // Stage A tile (16 rows x 3*FIN f16) into wave-private LDS.
        // Chunk id -> (row, 4-col); tensor select is a uniform compile-time loop: no divergence.
        if (full) {
#pragma unroll
            for (int t = 0; t < 3; ++t) {
                const float* __restrict__ Tp = Ts[t];
#pragma unroll
                for (int it = 0; it < FIN / 8; ++it) {
                    const int id  = lane + 32 * it;
                    const int row = id / (FIN / 4);
                    const int c4  = (id % (FIN / 4)) * 4;
                    v4f val = *(const v4f*)(Tp + (size_t)(tile * 16 + row) * FIN + c4);
                    v4h hv;
#pragma unroll
                    for (int j = 0; j < 4; ++j) hv[j] = (_Float16)val[j];
                    *(v4h*)(&sAw[row * KPAD + t * FIN + c4]) = hv;
                }
            }
        } else {
#pragma unroll
            for (int t = 0; t < 3; ++t) {
                const float* __restrict__ Tp = Ts[t];
#pragma unroll
                for (int it = 0; it < FIN / 8; ++it) {
                    const int id  = lane + 32 * it;
                    const int row = id / (FIN / 4);
                    const int c4  = (id % (FIN / 4)) * 4;
                    v4f val = {};
                    if (tile * 16 + row < nrows)
                        val = *(const v4f*)(Tp + (size_t)(tile * 16 + row) * FIN + c4);
                    v4h hv;
#pragma unroll
                    for (int j = 0; j < 4; ++j) hv[j] = (_Float16)val[j];
                    *(v4h*)(&sAw[row * KPAD + t * FIN + c4]) = hv;
                }
            }
        }

        v8f acc[NTILES];
#pragma unroll
        for (int nt = 0; nt < NTILES; ++nt) {
            v8f z = {0.f, 0.f, 0.f, 0.f, 0.f, 0.f, 0.f, 0.f};
            acc[nt] = z;
        }

#pragma unroll
        for (int ks = 0; ks < KSTEPS; ++ks) {
            // A fragment, contiguous k order used here (same as B): elem i, lane l:
            //   row = l&15, K = i + 16*(l>=16)
            v16h afrag;
#pragma unroll
            for (int i = 0; i < 16; ++i) {
                const int k = ks * 32 + hi * 16 + i;
                afrag[i] = sAw[lrow * KPAD + k];
            }
#pragma unroll
            for (int nt = 0; nt < NTILES; ++nt) {
                acc[nt] = __builtin_amdgcn_wmma_f32_16x16x32_f16(
                    false, afrag, false, Bf[ks * NTILES + nt], (short)0, acc[nt],
                    false, false);
            }
        }

        // Epilogue: C/D layout -> row = j + 8*(l>=16), col = l&15. Bias + ReLU.
        if (full) {
#pragma unroll
            for (int nt = 0; nt < NTILES; ++nt) {
                // One base address; 8 stores with compile-time immediate offsets j*FOUT*4.
                float* __restrict__ outp =
                    out + (size_t)(tile * 16 + hi * 8) * FOUT + nt * 16 + lrow;
#pragma unroll
                for (int j = 0; j < 8; ++j)
                    outp[(size_t)j * FOUT] = fmaxf(acc[nt][j] + bv[nt], 0.0f);
            }
        } else {
#pragma unroll
            for (int nt = 0; nt < NTILES; ++nt) {
                const int c = nt * 16 + lrow;
#pragma unroll
                for (int j = 0; j < 8; ++j) {
                    const int rr = tile * 16 + j + hi * 8;
                    if (rr < nrows)
                        out[(size_t)rr * FOUT + c] = fmaxf(acc[nt][j] + bv[nt], 0.0f);
                }
            }
        }
    }
}

// Layer 1: Fin=1, Fout=16, skip path folded into W1[0]. One thread per row, float4 stores.
__global__ void layer1_kernel(const float* __restrict__ x, const float* __restrict__ T1,
                              const float* __restrict__ T2, const float* __restrict__ W1,
                              const float* __restrict__ Wskip, const float* __restrict__ b1,
                              float* __restrict__ H1, int nrows) {
    const int r = blockIdx.x * blockDim.x + threadIdx.x;
    if (r >= nrows) return;
    const float t0 = x[r], t1 = T1[r], t2 = T2[r];
#pragma unroll
    for (int c = 0; c < 4; ++c) {
        v4f w0 = *(const v4f*)(W1 + 4 * c) + *(const v4f*)(Wskip + 4 * c);
        v4f h  = t0 * w0
               + t1 * (*(const v4f*)(W1 + 16 + 4 * c))
               + t2 * (*(const v4f*)(W1 + 32 + 4 * c))
               + *(const v4f*)(b1 + 4 * c);
#pragma unroll
        for (int j = 0; j < 4; ++j) h[j] = h[j] > 0.0f ? h[j] : 0.0f;
        *(v4f*)(H1 + (size_t)r * 16 + 4 * c) = h;
    }
}

// Final K=1 projection 16 -> 1 (no ReLU). One thread per row, float4 loads.
__global__ void final_kernel(const float* __restrict__ H3, const float* __restrict__ Wf,
                             const float* __restrict__ bf, float* __restrict__ out, int nrows) {
    const int r = blockIdx.x * blockDim.x + threadIdx.x;
    if (r >= nrows) return;
    float acc = bf[0];
#pragma unroll
    for (int c = 0; c < 4; ++c) {
        v4f h = *(const v4f*)(H3 + (size_t)r * 16 + 4 * c);
        v4f w = *(const v4f*)(Wf + 4 * c);
#pragma unroll
        for (int j = 0; j < 4; ++j) acc += h[j] * w[j];
    }
    out[r] = acc;
}

// fp32 stacked weights [K, N] -> f16 padded [KPAD, N] (zero rows for k >= K).
__global__ void convert_pad_kernel(const float* __restrict__ W, _Float16* __restrict__ Wh,
                                   int K, int KPAD, int N) {
    const int gid = blockIdx.x * blockDim.x + threadIdx.x;
    if (gid >= KPAD * N) return;
    const int k = gid / N;
    Wh[gid] = (k < K) ? (_Float16)W[gid] : (_Float16)0.0f;
}

extern "C" void kernel_launch(void* const* d_in, const int* in_sizes, int n_in,
                              void* d_out, int out_size, void* d_ws, size_t ws_size,
                              hipStream_t stream) {
    const float* x     = (const float*)d_in[0];
    const int*   rows  = (const int*)d_in[1];
    const int*   cols  = (const int*)d_in[2];
    const float* vals  = (const float*)d_in[3];
    const float* W1    = (const float*)d_in[4];
    const float* b1    = (const float*)d_in[5];
    const float* Wskip = (const float*)d_in[6];
    const float* W2    = (const float*)d_in[7];   // [3,16,64] == stacked [48,64]
    const float* b2    = (const float*)d_in[8];
    const float* W3    = (const float*)d_in[9];   // [3,64,16] == stacked [192,16]
    const float* b3    = (const float*)d_in[10];
    const float* Wf    = (const float*)d_in[11];
    const float* bf    = (const float*)d_in[12];

    const int BV = in_sizes[0];        // B*V (F=1 on input)
    const int V  = BV / BATCH;
    const int E  = in_sizes[1];

    // Workspace layout (fp32): R1[BV,64] R2[BV,64] R3[BV,64] R4[BV,16] + f16 weights.
    float* R1 = (float*)d_ws;
    float* R2 = R1 + (size_t)BV * 64;
    float* R3 = R2 + (size_t)BV * 64;
    float* R4 = R3 + (size_t)BV * 64;
    _Float16* Wh2 = (_Float16*)(R4 + (size_t)BV * 16);   // [64,64]
    _Float16* Wh3 = Wh2 + 64 * 64;                       // [192,16]

    convert_pad_kernel<<<(64 * 64 + 255) / 256, 256, 0, stream>>>(W2, Wh2, 48, 64, 64);
    convert_pad_kernel<<<(192 * 16 + 255) / 256, 256, 0, stream>>>(W3, Wh3, 192, 192, 16);

    const int spmm_blocks = (V * 32 + 255) / 256;
    const int ntiles = (BV + 15) / 16;
    const int gemm_blocks = (ntiles + 3) / 4;

    // ----- Layer 1 (Fin=1 -> 16, with skip) -----
    spmm_kernel<1><<<spmm_blocks, 256, 0, stream>>>(rows, cols, vals, E, V, x,  nullptr, 1.0f, R2); // T1
    spmm_kernel<1><<<spmm_blocks, 256, 0, stream>>>(rows, cols, vals, E, V, R2, x,       2.0f, R3); // T2
    layer1_kernel<<<(BV + 255) / 256, 256, 0, stream>>>(x, R2, R3, W1, Wskip, b1, R4, BV);          // H1

    // ----- Layer 2 (16 -> 64) -----
    spmm_kernel<16><<<spmm_blocks, 256, 0, stream>>>(rows, cols, vals, E, V, R4, nullptr, 1.0f, R2);
    spmm_kernel<16><<<spmm_blocks, 256, 0, stream>>>(rows, cols, vals, E, V, R2, R4,      2.0f, R3);
    cheb_wmma_kernel<16, 64, 2, 4><<<gemm_blocks, 128, 0, stream>>>(R4, R2, R3, Wh2, b2, R1, BV);   // H2

    // ----- Layer 3 (64 -> 16) -----
    spmm_kernel<64><<<spmm_blocks, 256, 0, stream>>>(rows, cols, vals, E, V, R1, nullptr, 1.0f, R2);
    spmm_kernel<64><<<spmm_blocks, 256, 0, stream>>>(rows, cols, vals, E, V, R2, R1,      2.0f, R3);
    cheb_wmma_kernel<64, 16, 6, 1><<<gemm_blocks, 128, 0, stream>>>(R1, R2, R3, Wh3, b3, R4, BV);   // H3

    // ----- Final (16 -> 1) -----
    final_kernel<<<(BV + 255) / 256, 256, 0, stream>>>(R4, Wf, bf, (float*)d_out, BV);
}
